// GroupedQueryAttention_5488968204632
// MI455X (gfx1250) — matrix operand order repaired and checked by reading, not compiled, not run on hardware
//
#include <hip/hip_runtime.h>
#include <cstdint>
#include <cstddef>

// ---------------------------------------------------------------- constants
#define L_SEQ 2048
#define HID   2048
#define NH    32
#define NKV   8
#define HD    64
#define DKV   (NKV * HD)   // 512

typedef unsigned short u16;
typedef __attribute__((ext_vector_type(16))) __bf16 bf16x16;
typedef __attribute__((ext_vector_type(8)))  __bf16 bf16x8;
typedef __attribute__((ext_vector_type(8)))  float  floatx8;
typedef __attribute__((ext_vector_type(4))) unsigned int v4u;
typedef __attribute__((ext_vector_type(8))) int          v8i;
typedef __attribute__((ext_vector_type(4))) int          v4i;

union ABfrag { bf16x16 v; bf16x8 h[2]; };

__device__ __forceinline__ u16 f2bf(float f) {
  union { float f; unsigned int u; } c; c.f = f;
  unsigned int u = c.u;
  return (u16)((u + 0x7FFFu + ((u >> 16) & 1u)) >> 16);   // RNE
}

__device__ __forceinline__ floatx8 wmma_bf16(bf16x16 a, bf16x16 b, floatx8 c) {
  // D = A(16x32 bf16) * B(32x16 bf16) + C(16x16 f32)
  return __builtin_amdgcn_wmma_f32_16x16x32_bf16(false, a, false, b,
                                                 (short)0, c, false, false);
}

// TDM: async-load a 16-row x 32-col bf16 tile (row stride dim0 elements) from
// global to LDS, per CDNA5 D# layout (CDNA5 ISA).
// This toolchain exposes the 6-arg builtin form:
//   (v4u group0, v8i group1, v4i group2, v4i group3, v8i extra, i32 cpol)
__device__ __forceinline__ void tdm_load_2d(unsigned lds_off,
                                            unsigned long long gaddr,
                                            unsigned dim0, unsigned dim1) {
  v4u g0;
  g0[0] = 1u;                                   // count=1, user descriptor
  g0[1] = lds_off;                              // lds_addr (bytes)
  g0[2] = (unsigned)gaddr;                      // global_addr[31:0]
  g0[3] = (unsigned)(gaddr >> 32) | (2u << 30); // global_addr[56:32] | type=2
  v8i g1;
  g1[0] = (int)(1u << 16);                      // wg_mask=0, data_size=2B
  g1[1] = (int)((dim0 & 0xFFFFu) << 16);        // tensor_dim0[15:0]
  g1[2] = (int)(((dim0 >> 16) & 0xFFFFu) | ((dim1 & 0xFFFFu) << 16));
  g1[3] = (int)(((dim1 >> 16) & 0xFFFFu) | (32u << 16)); // tile_dim0=32
  g1[4] = 16;                                   // tile_dim1=16, tile_dim2=0
  g1[5] = (int)dim0;                            // tensor_dim0_stride
  g1[6] = 0;
  g1[7] = 0;
  v4i z4 = {0, 0, 0, 0};
  v8i z8 = {0, 0, 0, 0, 0, 0, 0, 0};
  __builtin_amdgcn_tensor_load_to_lds(g0, g1, z4, z4, z8, 0);
}

#define FZERO8 {0.f,0.f,0.f,0.f,0.f,0.f,0.f,0.f}

// ---------------------------------------------------------------- fp32 -> bf16
__global__ void __launch_bounds__(256)
f32_to_bf16_k(const float* __restrict__ in, u16* __restrict__ out, int n) {
  int i = blockIdx.x * 256 + threadIdx.x;
  if (i < n) out[i] = f2bf(in[i]);
}

// ---------------------------------------------------------------- RMSNorm -> bf16
__global__ void __launch_bounds__(256)
rmsnorm_bf16(const float* __restrict__ x, const float* __restrict__ w,
             u16* __restrict__ h) {
  __shared__ float red[256];
  const int row = blockIdx.x, tid = threadIdx.x;
  const float4* xr = (const float4*)(x + (size_t)row * HID);
  float4 a = xr[tid];
  float4 b = xr[tid + 256];
  float ss = a.x*a.x + a.y*a.y + a.z*a.z + a.w*a.w
           + b.x*b.x + b.y*b.y + b.z*b.z + b.w*b.w;
  red[tid] = ss; __syncthreads();
  for (int s = 128; s > 0; s >>= 1) {
    if (tid < s) red[tid] += red[tid + s];
    __syncthreads();
  }
  const float scale = rsqrtf(red[0] * (1.0f / HID) + 1e-6f);
  u16* hr = h + (size_t)row * HID;
  const int i0 = tid * 4, i1 = 1024 + tid * 4;
  hr[i0+0] = f2bf(a.x * scale * w[i0+0]);
  hr[i0+1] = f2bf(a.y * scale * w[i0+1]);
  hr[i0+2] = f2bf(a.z * scale * w[i0+2]);
  hr[i0+3] = f2bf(a.w * scale * w[i0+3]);
  hr[i1+0] = f2bf(b.x * scale * w[i1+0]);
  hr[i1+1] = f2bf(b.y * scale * w[i1+1]);
  hr[i1+2] = f2bf(b.z * scale * w[i1+2]);
  hr[i1+3] = f2bf(b.w * scale * w[i1+3]);
}

// ---------------------------------------------------------------- WMMA GEMM
// C[M,N](f32) = A[M,K](bf16) * W[N,K](bf16)^T (+residual).
// Block = 16(M) x 512(N): 8 waves x (16x64) tiles sharing one A tile.
// A tile staged to LDS by the Tensor Data Mover (double-buffered); B fragments
// ping-pong between two register sets (manual 2x unroll, no copies).
// K must be a multiple of 64.
__global__ void __launch_bounds__(256)
gemm_bf16_wmma(const u16* __restrict__ A, const u16* __restrict__ W,
               const float* __restrict__ residual, float* __restrict__ C,
               int M, int N, int K) {
  __shared__ u16 lds_a[2][16 * 32];
  const int wid  = threadIdx.x >> 5;
  const int lane = threadIdx.x & 31;
  const int half = lane >> 4;
  const int hm   = lane & 15;
  const int nblk = N >> 9;
  const int m0 = (int)(blockIdx.x / nblk) << 4;
  const int n0 = ((int)(blockIdx.x % nblk) << 9) + (wid << 6);

  const unsigned lds_off0 = (unsigned)(uintptr_t)&lds_a[0][0];
  const unsigned lds_off1 = (unsigned)(uintptr_t)&lds_a[1][0];
  const unsigned long long abase =
      (unsigned long long)(uintptr_t)(A + (size_t)m0 * K);

  // prologue: TDM-load first A tile into buffer 0
  if (wid == 0) tdm_load_2d(lds_off0, abase, (unsigned)K, (unsigned)M);
  __builtin_amdgcn_s_wait_tensorcnt(0);   // no-op on waves that issued nothing
  __syncthreads();

  const u16* w0 = W + (size_t)(n0      + hm) * K + half * 16;
  const u16* w1 = W + (size_t)(n0 + 16 + hm) * K + half * 16;
  const u16* w2 = W + (size_t)(n0 + 32 + hm) * K + half * 16;
  const u16* w3 = W + (size_t)(n0 + 48 + hm) * K + half * 16;

  bf16x16 b0 = *(const bf16x16*)(w0);
  bf16x16 b1 = *(const bf16x16*)(w1);
  bf16x16 b2 = *(const bf16x16*)(w2);
  bf16x16 b3 = *(const bf16x16*)(w3);
  bf16x16 c0, c1, c2, c3;

  floatx8 acc0 = FZERO8, acc1 = FZERO8, acc2 = FZERO8, acc3 = FZERO8;

  for (int k0 = 0; k0 < K; k0 += 64) {
    // ======== phase 0: compute on lds_a[0] + b-set; fetch k0+32 ========
    const int km = k0 + 32;                       // always < K (K % 64 == 0)
    if (wid == 0)
      tdm_load_2d(lds_off1, abase + (size_t)km * 2, (unsigned)K, (unsigned)M);
    c0 = *(const bf16x16*)(w0 + km);
    c1 = *(const bf16x16*)(w1 + km);
    c2 = *(const bf16x16*)(w2 + km);
    c3 = *(const bf16x16*)(w3 + km);
    {
      ABfrag a;
      const u16* ap = &lds_a[0][0];
      a.h[0] = *(const bf16x8*)(ap + hm * 32 +     half * 16);
      a.h[1] = *(const bf16x8*)(ap + hm * 32 + 8 + half * 16);
      acc0 = wmma_bf16(a.v, b0, acc0);
      acc1 = wmma_bf16(a.v, b1, acc1);
      acc2 = wmma_bf16(a.v, b2, acc2);
      acc3 = wmma_bf16(a.v, b3, acc3);
    }
    __builtin_amdgcn_s_wait_tensorcnt(0);
    __syncthreads();

    // ======== phase 1: compute on lds_a[1] + c-set; fetch k0+64 ========
    const int kn = k0 + 64;
    if (wid == 0 && kn < K)
      tdm_load_2d(lds_off0, abase + (size_t)kn * 2, (unsigned)K, (unsigned)M);
    if (kn < K) {
      b0 = *(const bf16x16*)(w0 + kn);
      b1 = *(const bf16x16*)(w1 + kn);
      b2 = *(const bf16x16*)(w2 + kn);
      b3 = *(const bf16x16*)(w3 + kn);
      __builtin_prefetch(w0 + kn + 32, 0, 1);   // global_prefetch_b8
      __builtin_prefetch(w2 + kn + 32, 0, 1);
    }
    {
      ABfrag a;
      const u16* ap = &lds_a[1][0];
      a.h[0] = *(const bf16x8*)(ap + hm * 32 +     half * 16);
      a.h[1] = *(const bf16x8*)(ap + hm * 32 + 8 + half * 16);
      acc0 = wmma_bf16(a.v, c0, acc0);
      acc1 = wmma_bf16(a.v, c1, acc1);
      acc2 = wmma_bf16(a.v, c2, acc2);
      acc3 = wmma_bf16(a.v, c3, acc3);
    }
    __builtin_amdgcn_s_wait_tensorcnt(0);
    __syncthreads();
  }

  #pragma unroll
  for (int r = 0; r < 8; ++r) {
    const int row = m0 + r + 8 * half;
    float* crow = C + (size_t)row * N + n0 + hm;
    float v0 = acc0[r], v1 = acc1[r], v2 = acc2[r], v3 = acc3[r];
    if (residual) {
      const float* rr = residual + (size_t)row * N + n0 + hm;
      v0 += rr[0]; v1 += rr[16]; v2 += rr[32]; v3 += rr[48];
    }
    crow[0] = v0; crow[16] = v1; crow[32] = v2; crow[48] = v3;
  }
}

// ---------------------------------------------------------------- RoPE scatter
// Qf32[L][NH*64] -> Qb[NH][L][64] (bf16, roped)
__global__ void __launch_bounds__(256)
rope_q_k(const float* __restrict__ Qf, const float* __restrict__ rc,
         const float* __restrict__ rs, u16* __restrict__ Qb) {
  const int idx = blockIdx.x * 256 + threadIdx.x;      // L*NH*32 threads
  const int pos = idx >> 10;
  const int rem = idx & 1023;
  const int head = rem >> 5;
  const int i = rem & 31;
  const float c = rc[pos * HD + 2 * i];
  const float s = rs[pos * HD + 2 * i];
  const float x1 = Qf[(size_t)pos * HID + head * HD + 2 * i];
  const float x2 = Qf[(size_t)pos * HID + head * HD + 2 * i + 1];
  u16* o = Qb + ((size_t)head * L_SEQ + pos) * HD + 2 * i;
  o[0] = f2bf(x1 * c - x2 * s);
  o[1] = f2bf(x2 * c + x1 * s);
}

// Kf32[L][512] -> Kb[NKV][L][64] (bf16, roped)
__global__ void __launch_bounds__(256)
rope_kv_k(const float* __restrict__ Kf, const float* __restrict__ rc,
          const float* __restrict__ rs, u16* __restrict__ Kb) {
  const int idx = blockIdx.x * 256 + threadIdx.x;      // L*NKV*32 threads
  const int pos = idx >> 8;
  const int rem = idx & 255;
  const int kv = rem >> 5;
  const int i = rem & 31;
  const float c = rc[pos * HD + 2 * i];
  const float s = rs[pos * HD + 2 * i];
  const float x1 = Kf[(size_t)pos * DKV + kv * HD + 2 * i];
  const float x2 = Kf[(size_t)pos * DKV + kv * HD + 2 * i + 1];
  u16* o = Kb + ((size_t)kv * L_SEQ + pos) * HD + 2 * i;
  o[0] = f2bf(x1 * c - x2 * s);
  o[1] = f2bf(x2 * c + x1 * s);
}

// Vf32[L][512] -> Vt[NKV][64][L] (bf16, transposed for PV B-fragments)
__global__ void __launch_bounds__(256)
v_transpose_k(const float* __restrict__ Vf, u16* __restrict__ Vt) {
  const int idx = blockIdx.x * 256 + threadIdx.x;      // L*DKV threads
  const int pos = idx >> 9;
  const int c = idx & 511;
  const int kv = c >> 6, d = c & 63;
  Vt[((size_t)kv * HD + d) * L_SEQ + pos] = f2bf(Vf[(size_t)pos * DKV + c]);
}

// ---------------------------------------------------------------- flash attention
// One wave per (head, 16-query block). 32 keys per step, online softmax.
__global__ void __launch_bounds__(256)
flash_attn_wmma(const u16* __restrict__ Qb, const u16* __restrict__ Kb,
                const u16* __restrict__ Vt, u16* __restrict__ attn) {
  __shared__ u16 lds_p[8 * 16 * 32];                   // per-wave 16x32 P tile
  const int wid  = threadIdx.x >> 5;
  const int lane = threadIdx.x & 31;
  const int half = lane >> 4;
  const int hm   = lane & 15;
  const int task = blockIdx.x * 8 + wid;
  const int head = task >> 7;                          // 128 q-blocks per head
  const int qb   = task & 127;
  const int kv   = head >> 2;                          // GQA group of 4
  const int m0   = qb << 4;
  u16* pb = lds_p + wid * 512;

  // Q fragments for d=0..31 and d=32..63, invariant over key blocks
  const u16* qrow = Qb + ((size_t)head * L_SEQ + (m0 + hm)) * HD;
  ABfrag aq0, aq1;
  aq0.h[0] = *(const bf16x8*)(qrow +  0 + half * 16);
  aq0.h[1] = *(const bf16x8*)(qrow +  8 + half * 16);
  aq1.h[0] = *(const bf16x8*)(qrow + 32 + half * 16);
  aq1.h[1] = *(const bf16x8*)(qrow + 40 + half * 16);

  floatx8 o0 = FZERO8, o1 = FZERO8, o2 = FZERO8, o3 = FZERO8;
  float mrow[8], lrow[8];
  #pragma unroll
  for (int r = 0; r < 8; ++r) { mrow[r] = -1e30f; lrow[r] = 0.f; }

  const int nkb = (m0 + 47) >> 5;                      // causal bound
  for (int kb = 0; kb < nkb; ++kb) {
    const int kbase = kb << 5;
    // ---- S = Q K^T (16x32), 4 WMMAs over d=64
    floatx8 s0 = FZERO8, s1 = FZERO8;
    {
      const u16* k0p = Kb + ((size_t)kv * L_SEQ + (kbase + hm)) * HD;
      const u16* k1p = k0p + 16 * HD;
      bf16x16 b;
      b = *(const bf16x16*)(k0p +      half * 16);  s0 = wmma_bf16(aq0.v, b, s0);
      b = *(const bf16x16*)(k0p + 32 + half * 16);  s0 = wmma_bf16(aq1.v, b, s0);
      b = *(const bf16x16*)(k1p +      half * 16);  s1 = wmma_bf16(aq0.v, b, s1);
      b = *(const bf16x16*)(k1p + 32 + half * 16);  s1 = wmma_bf16(aq1.v, b, s1);
    }
    // ---- V fragments issued early: global loads overlap the softmax ALU work
    const u16* vtb = Vt + (size_t)kv * HD * L_SEQ + kbase + half * 16;
    bf16x16 bv0 = *(const bf16x16*)(vtb + (size_t)(     hm) * L_SEQ);
    bf16x16 bv1 = *(const bf16x16*)(vtb + (size_t)(16 + hm) * L_SEQ);
    bf16x16 bv2 = *(const bf16x16*)(vtb + (size_t)(32 + hm) * L_SEQ);
    bf16x16 bv3 = *(const bf16x16*)(vtb + (size_t)(48 + hm) * L_SEQ);

    // ---- scale + causal mask + online softmax, write P tile to LDS (bf16)
    #pragma unroll
    for (int r = 0; r < 8; ++r) {
      const int row = m0 + r + 8 * half;
      const int c0 = kbase + hm;
      float v0 = (c0      <= row) ? s0[r] * 0.125f : -1e30f;
      float v1 = (c0 + 16 <= row) ? s1[r] * 0.125f : -1e30f;
      float vm = fmaxf(v0, v1);
      #pragma unroll
      for (int sh = 8; sh >= 1; sh >>= 1) vm = fmaxf(vm, __shfl_xor(vm, sh, 32));
      const float nm   = fmaxf(mrow[r], vm);
      const float corr = __expf(mrow[r] - nm);
      const float p0 = __expf(v0 - nm);
      const float p1 = __expf(v1 - nm);
      float rsum = p0 + p1;
      #pragma unroll
      for (int sh = 8; sh >= 1; sh >>= 1) rsum += __shfl_xor(rsum, sh, 32);
      mrow[r] = nm;
      lrow[r] = lrow[r] * corr + rsum;
      o0[r] *= corr; o1[r] *= corr; o2[r] *= corr; o3[r] *= corr;
      const int prow = (r + 8 * half) * 32;
      pb[prow +      hm] = f2bf(p0);
      pb[prow + 16 + hm] = f2bf(p1);
    }
    // DS ops are in-order per wave; fence the C-layout -> A-layout round trip.
    asm volatile("s_wait_dscnt 0" ::: "memory");
    ABfrag ap;
    ap.h[0] = *(const bf16x8*)(pb + hm * 32 +     half * 16);
    ap.h[1] = *(const bf16x8*)(pb + hm * 32 + 8 + half * 16);
    // ---- O += P V (16x64), 4 WMMAs
    o0 = wmma_bf16(ap.v, bv0, o0);
    o1 = wmma_bf16(ap.v, bv1, o1);
    o2 = wmma_bf16(ap.v, bv2, o2);
    o3 = wmma_bf16(ap.v, bv3, o3);
  }

  // ---- normalize and write token-major bf16 [L][NH*64]
  #pragma unroll
  for (int r = 0; r < 8; ++r) {
    const float inv = 1.0f / lrow[r];
    const int row = m0 + r + 8 * half;
    u16* orow = attn + (size_t)row * HID + head * HD + hm;
    orow[0]  = f2bf(o0[r] * inv);
    orow[16] = f2bf(o1[r] * inv);
    orow[32] = f2bf(o2[r] * inv);
    orow[48] = f2bf(o3[r] * inv);
  }
}

// ---------------------------------------------------------------- launch
extern "C" void kernel_launch(void* const* d_in, const int* in_sizes, int n_in,
                              void* d_out, int out_size, void* d_ws, size_t ws_size,
                              hipStream_t stream) {
  (void)in_sizes; (void)n_in; (void)out_size; (void)ws_size;
  const float* x      = (const float*)d_in[0];
  // d_in[1] = mask (causal mask applied analytically in-kernel)
  const float* norm_w = (const float*)d_in[2];
  const float* Wq     = (const float*)d_in[3];
  const float* Wk     = (const float*)d_in[4];
  const float* Wv     = (const float*)d_in[5];
  const float* Wo     = (const float*)d_in[6];
  const float* rc     = (const float*)d_in[7];
  const float* rs     = (const float*)d_in[8];
  float* out = (float*)d_out;

  // workspace layout (bytes); total 72 MB
  char* w = (char*)d_ws;
  const size_t MB = 1024 * 1024;
  u16*   h_b  = (u16*)(w + 0 * MB);     // [2048][2048] bf16   8 MB
  u16*   Wqb  = (u16*)(w + 8 * MB);     // [2048][2048] bf16   8 MB
  u16*   Wkb  = (u16*)(w + 16 * MB);    // [512][2048]  bf16   2 MB
  u16*   Wvb  = (u16*)(w + 18 * MB);    // [512][2048]  bf16   2 MB
  u16*   Wob  = (u16*)(w + 20 * MB);    // [2048][2048] bf16   8 MB
  float* Qf   = (float*)(w + 28 * MB);  // [2048][2048] f32   16 MB
  float* Kf   = (float*)(w + 44 * MB);  // [2048][512]  f32    4 MB
  float* Vf   = (float*)(w + 48 * MB);  // [2048][512]  f32    4 MB
  u16*   Qb   = (u16*)(w + 52 * MB);    // [32][2048][64] bf16 8 MB
  u16*   Kb   = (u16*)(w + 60 * MB);    // [8][2048][64] bf16  2 MB
  u16*   Vt   = (u16*)(w + 62 * MB);    // [8][64][2048] bf16  2 MB
  u16*   At   = (u16*)(w + 64 * MB);    // [2048][2048] bf16   8 MB

  // 1) weights -> bf16
  f32_to_bf16_k<<<(HID * HID) / 256, 256, 0, stream>>>(Wq, Wqb, HID * HID);
  f32_to_bf16_k<<<(DKV * HID) / 256, 256, 0, stream>>>(Wk, Wkb, DKV * HID);
  f32_to_bf16_k<<<(DKV * HID) / 256, 256, 0, stream>>>(Wv, Wvb, DKV * HID);
  f32_to_bf16_k<<<(HID * HID) / 256, 256, 0, stream>>>(Wo, Wob, HID * HID);

  // 2) RMSNorm -> bf16 h
  rmsnorm_bf16<<<L_SEQ, 256, 0, stream>>>(x, norm_w, h_b);

  // 3) Q/K/V projections (WMMA + TDM-staged A tiles)
  gemm_bf16_wmma<<<(L_SEQ / 16) * (HID / 512), 256, 0, stream>>>(
      h_b, Wqb, nullptr, Qf, L_SEQ, HID, HID);
  gemm_bf16_wmma<<<(L_SEQ / 16) * (DKV / 512), 256, 0, stream>>>(
      h_b, Wkb, nullptr, Kf, L_SEQ, DKV, HID);
  gemm_bf16_wmma<<<(L_SEQ / 16) * (DKV / 512), 256, 0, stream>>>(
      h_b, Wvb, nullptr, Vf, L_SEQ, DKV, HID);

  // 4) RoPE + layout transforms -> bf16
  rope_q_k<<<(L_SEQ * NH * 32) / 256, 256, 0, stream>>>(Qf, rc, rs, Qb);
  rope_kv_k<<<(L_SEQ * NKV * 32) / 256, 256, 0, stream>>>(Kf, rc, rs, Kb);
  v_transpose_k<<<(L_SEQ * DKV) / 256, 256, 0, stream>>>(Vf, Vt);

  // 5) flash attention (WMMA)
  flash_attn_wmma<<<(NH * (L_SEQ / 16)) / 8, 256, 0, stream>>>(Qb, Kb, Vt, At);

  // 6) output projection + residual (WMMA)
  gemm_bf16_wmma<<<(L_SEQ / 16) * (HID / 512), 256, 0, stream>>>(
      At, Wob, x, out, L_SEQ, HID, HID);
}
